// RWKV_TimeMix_16028817949581
// MI455X (gfx1250) — matrix operand order repaired and checked by reading, not compiled, not run on hardware
//
#include <hip/hip_runtime.h>

// ---------------------------------------------------------------------------
// RWKV TimeMix for MI455X (gfx1250, wave32, WMMA, async-to-LDS).
// B=8 T=1024 C=1024 H=16 head=64 CTX=1024.  ~86 GFLOP, ~150 MB intermediates:
// memory-bound => 16-bit intermediates (L2-resident), fused epilogues,
// b128-only data movement, 3-deep async pipelined GEMM staging (saddr form).
// ---------------------------------------------------------------------------

typedef __attribute__((ext_vector_type(16))) _Float16 v16h;
typedef __attribute__((ext_vector_type(16))) __bf16   v16bf;
typedef __attribute__((ext_vector_type(8)))  float    v8f;

#define NB 8
#define NT 1024
#define NC 1024
#define NH 16
#define HS 64
#define NCTX 1024
#define LDT 40   // LDS tile row stride in halfs (80 B, keeps 16 B alignment)

// ---------------------------------------------------- CDNA5 async-to-LDS copy
// SADDR form: mem = SGPR base + 32-bit VGPR offset. Per-thread offset is loop
// invariant; the per-tile advance is a uniform SALU pointer bump (no 64-bit
// VGPR adds, no WMMA->VALU hazard NOPs in the hot loop).
__device__ __forceinline__ void async_b128_s(unsigned lds_off, unsigned voff,
                                             const void* sbase) {
  asm volatile("global_load_async_to_lds_b128 %0, %1, %2"
               :: "v"(lds_off), "v"(voff), "s"(sbase) : "memory");
}
// Async loads complete in order: waiting ASYNCcnt<=N peels exactly the oldest
// issues while newer tiles keep streaming.
template <int N>
__device__ __forceinline__ void wait_async() {
  asm volatile("s_wait_asynccnt %0" :: "n"(N) : "memory");
}

// ---------------------------------------------------------------- elementwise

__global__ __launch_bounds__(256) void shift_to_f16(const float* __restrict__ x,
                                                    _Float16* __restrict__ xs,
                                                    int total) {
  int idx = blockIdx.x * 256 + threadIdx.x;
  if (idx >= total) return;
  int c = idx & (NC - 1);
  int t = (idx >> 10) & (NT - 1);
  float v;
  if (c < NC / 2) {                 // time_shift: first half shifted right by 1
    v = (t > 0) ? x[idx - NC] : 0.f;
  } else {
    v = x[idx];
  }
  xs[idx] = (_Float16)v;
}

// W (K,N) f32 -> Wt (N,K) f16, LDS-tiled so both sides stay coalesced
__global__ __launch_bounds__(256) void transpose_w_f16(const float* __restrict__ W,
                                                       _Float16* __restrict__ Wt,
                                                       int K, int N) {
  __shared__ float tile[32][33];
  const int bx = blockIdx.x * 32;   // N
  const int by = blockIdx.y * 32;   // K
  const int tx = threadIdx.x & 31;
  const int ty = threadIdx.x >> 5;  // 0..7
#pragma unroll
  for (int i = 0; i < 4; ++i) {
    int k = by + ty * 4 + i;
    tile[ty * 4 + i][tx] = W[(size_t)k * N + bx + tx];
  }
  __syncthreads();
#pragma unroll
  for (int i = 0; i < 4; ++i) {
    int n = bx + ty * 4 + i;
    Wt[(size_t)n * K + by + tx] = (_Float16)tile[tx][ty * 4 + i];
  }
}

// kvT[b][ch][u] = bf16( alpha[h,u] * k[b,u,ch] * v[b,u,ch] ), LDS-tiled transpose
__global__ __launch_bounds__(256) void kvT_prep(const float* __restrict__ k,
                                                const float* __restrict__ v,
                                                const float* __restrict__ alpha, // (H,1,CTX)
                                                __bf16* __restrict__ kvT) {
  __shared__ float tile[32][33];
  const int b = blockIdx.z;
  const int c0 = blockIdx.x * 32;
  const int u0 = blockIdx.y * 32;
  const int tx = threadIdx.x & 31;
  const int ty = threadIdx.x >> 5;
#pragma unroll
  for (int i = 0; i < 4; ++i) {
    int u = u0 + ty * 4 + i;
    size_t idx = ((size_t)b * NT + u) * NC + c0 + tx;
    int hh = (c0 + tx) >> 6;
    tile[ty * 4 + i][tx] = alpha[hh * NCTX + u] * k[idx] * v[idx];
  }
  __syncthreads();
#pragma unroll
  for (int i = 0; i < 4; ++i) {
    int ch = c0 + ty * 4 + i;
    kvT[((size_t)b * NC + ch) * NT + u0 + tx] = (__bf16)tile[tx][ty * 4 + i];
  }
}

__global__ __launch_bounds__(256) void cumsum_t(float* __restrict__ k) {
  int idx = blockIdx.x * 256 + threadIdx.x;  // b*NC + c
  if (idx >= NB * NC) return;
  int b = idx >> 10, c = idx & (NC - 1);
  const size_t base = (size_t)b * NT * NC + c;
  float s = 0.f;
  for (int t = 0; t < NT; ++t) {
    size_t o = base + (size_t)t * NC;
    s += k[o];
    k[o] = s;
  }
}

// Precompute the 16x32 causal decay tiles in WMMA A-fragment per-lane layout.
// Tile identity depends only on (h, d = (t0-u0)/16): 16*64 = 1024 tiles, 1 MB.
__global__ __launch_bounds__(32) void build_atiles(const float* __restrict__ tw, // (H,CTX)
                                                   __bf16* __restrict__ Atiles) {
  const int h = blockIdx.x >> 6;
  const int d = blockIdx.x & 63;
  const int lane = threadIdx.x;
  const int m = lane & 15, hi = lane >> 4;
  const int d0 = d * 16;
  v16bf frag;
#pragma unroll
  for (int j = 0; j < 16; ++j) {
    // contiguous K layout (matches kvT B operand): K = hi*16+j
    int kl = hi * 16 + j;
    int diff = d0 + m - kl;        // t - u
    float wv = (diff >= 0 && diff < NCTX) ? tw[h * NCTX + (NCTX - 1) - diff] : 0.f;
    frag[j] = (__bf16)wv;
  }
  *(v16bf*)(Atiles + ((size_t)blockIdx.x * 32 + lane) * 16) = frag;
}

// ------------------------------------------------------- dense GEMM (f16 WMMA)
// out[M,N] = act( A[M,K=1024] @ Wt[N,K]^T + bias ); 64x64x32 tiles, 8 waves,
// 3-deep async-to-LDS pipeline, all-b128 fragment loads, peeled final tile.
// At iteration s: tiles s+1, s+2 in flight; s_wait_asynccnt 2 == "tile s
// landed" (in-order completion), so each tile's DMA spans a full iteration.
// mode 0: exp(clip(y,-60,30))  1: identity  2: sigmoid  3: y*gamma[row&(T-1)]

__global__ __launch_bounds__(256) void gemm_f16_wmma(
    const _Float16* __restrict__ A, const _Float16* __restrict__ Wt,
    const float* __restrict__ bias, float* __restrict__ out,
    const float* __restrict__ gamma, int N, int mode) {
  constexpr int K = NC;           // 1024 for every GEMM in this pipeline
  constexpr int S = K / 32;       // 32 K-tiles
  __shared__ __align__(16) _Float16 As[3][64 * LDT];
  __shared__ __align__(16) _Float16 Bs[3][64 * LDT];

  const int tid = threadIdx.x;
  const int lane = tid & 31;
  const int wave = tid >> 5;      // 0..7
  const int wm = wave & 3;        // 16-row M sub-tile
  const int wn = wave >> 2;       // 0/1: 32-col N group
  const int m = lane & 15;
  const int hi = lane >> 4;

  const int n0 = blockIdx.x * 64;
  const int m0 = blockIdx.y * 64;

  // cooperative staging: 64 rows x 4 groups of 8 halfs (b128 per thread)
  const int ar = tid >> 2;
  const int ac = (tid & 3) * 8;
  const unsigned vA = (unsigned)(((m0 + ar) * K + ac) * sizeof(_Float16));
  const unsigned vB = (unsigned)(((n0 + ar) * K + ac) * sizeof(_Float16));
  const unsigned laOff = (unsigned)(unsigned long long)&As[0][ar * LDT + ac];
  const unsigned lbOff = (unsigned)(unsigned long long)&Bs[0][ar * LDT + ac];
  const unsigned bs = (unsigned)(64 * LDT * sizeof(_Float16));

  v8f acc0 = {}, acc1 = {};

  auto compute = [&](unsigned off) {
    union { uint4 q[2]; v16h v; } ua, ub0, ub1;
    const _Float16* arow =
        (const _Float16*)((const char*)&As[0][0] + off) + (wm * 16 + m) * LDT;
    ua.q[0] = *(const uint4*)(arow + hi * 16);
    ua.q[1] = *(const uint4*)(arow + hi * 16 + 8);
    const _Float16* bbase = (const _Float16*)((const char*)&Bs[0][0] + off);
    const _Float16* b0row = bbase + (wn * 32 + m) * LDT + hi * 16;
    const _Float16* b1row = bbase + (wn * 32 + 16 + m) * LDT + hi * 16;
    ub0.q[0] = *(const uint4*)(b0row);
    ub0.q[1] = *(const uint4*)(b0row + 8);
    ub1.q[0] = *(const uint4*)(b1row);
    ub1.q[1] = *(const uint4*)(b1row + 8);
    acc0 = __builtin_amdgcn_wmma_f32_16x16x32_f16(false, ua.v, false, ub0.v, (short)0, acc0, false, false);
    acc1 = __builtin_amdgcn_wmma_f32_16x16x32_f16(false, ua.v, false, ub1.v, (short)0, acc1, false, false);
  };

  // prime two stages (tiles 0 and 1)
  const char* gAt = (const char*)A;
  const char* gBt = (const char*)Wt;
  async_b128_s(laOff, vA, gAt);
  async_b128_s(lbOff, vB, gBt);
  async_b128_s(laOff + bs, vA, gAt + 64);
  async_b128_s(lbOff + bs, vB, gBt + 64);
  gAt += 128;                     // next tile to issue: tile 2
  gBt += 128;

  unsigned o0 = 0, o1 = bs, o2 = 2 * bs;   // LDS slots for tiles s, s+1, s+2
  for (int s = 0; s < S - 1; ++s) {
    wait_async<2>();              // tile s landed; tile s+1 still streaming
    __syncthreads();
    if (s + 2 < S) {
      async_b128_s(laOff + o2, vA, gAt);
      async_b128_s(lbOff + o2, vB, gBt);
      gAt += 64;
      gBt += 64;
    }
    compute(o0);
    unsigned tmp = o0; o0 = o1; o1 = o2; o2 = tmp;  // rotate slots
  }
  wait_async<0>();                // final tile: drain
  __syncthreads();
  compute(o0);

  // C/D: VGPR r -> rows r (lanes 0-15) / r+8 (lanes 16-31), N = lane&15
#pragma unroll
  for (int r = 0; r < 8; ++r) {
    int row = m0 + wm * 16 + hi * 8 + r;
    int c0 = n0 + wn * 32 + m;
    int c1 = c0 + 16;
    float y0 = acc0[r] + bias[c0];
    float y1 = acc1[r] + bias[c1];
    if (mode == 0) {
      y0 = __expf(fminf(fmaxf(y0, -60.f), 30.f));
      y1 = __expf(fminf(fmaxf(y1, -60.f), 30.f));
    } else if (mode == 2) {
      y0 = 1.f / (1.f + __expf(-y0));
      y1 = 1.f / (1.f + __expf(-y1));
    } else if (mode == 3) {
      float g = gamma[row & (NT - 1)];
      y0 *= g; y1 *= g;
    }
    out[(size_t)row * N + c0] = y0;
    out[(size_t)row * N + c1] = y1;
  }
}

// ------------------------------------------------- causal decay matmul (bf16)
// Block = (b, h, 16-row t-tile); 4 waves = 4 column segments of head 64.
// Inner loop: 32 B A-tile load (precomputed fragments) + 32 B kvT load + WMMA.
__global__ __launch_bounds__(128) void wkv_wmma(
    const __bf16* __restrict__ Atiles,    // [H*64][32 lanes][16]
    const float* __restrict__ time_beta,  // (H, CTX)
    const __bf16* __restrict__ kvT,       // (B*NC, NT)
    const float* __restrict__ rsig,       // (B,T,1024)
    const float* __restrict__ sumk,       // (B,T,1024)
    _Float16* __restrict__ rwkv16) {      // (B,T,1024)
  const int bid = blockIdx.x;
  const int tile = bid & 63;
  const int h = (bid >> 6) & (NH - 1);
  const int b = bid >> 10;
  const int tid = threadIdx.x;
  const int lane = tid & 31;
  const int wave = tid >> 5;              // c segment 0..3
  const int m = lane & 15;
  const int hi = lane >> 4;

  const int t0 = tile * 16;
  const int usteps = (t0 + 47) >> 5;      // causal: only u-tiles with u <= t

  const int ch = h * HS + wave * 16 + m;
  const __bf16* kvrow = kvT + ((size_t)b * NC + ch) * NT + hi * 16;
  const __bf16* abase = Atiles + (((size_t)h * 64) * 32 + lane) * 16;

  v8f acc = {};
#pragma unroll 2
  for (int s = 0; s < usteps; ++s) {
    const int u0 = s * 32;
    const int d = (t0 - u0) >> 4;         // tile selector: 0..63
    v16bf a  = *(const v16bf*)(abase + (size_t)d * (32 * 16));
    v16bf bf = *(const v16bf*)(kvrow + u0);
    acc = __builtin_amdgcn_wmma_f32_16x16x32_bf16(false, a, false, bf, (short)0, acc, false, false);
  }

#pragma unroll
  for (int r = 0; r < 8; ++r) {
    int tt = t0 + hi * 8 + r;
    size_t idx = ((size_t)b * NT + tt) * NC + ch;
    float bet = time_beta[h * NCTX + tt];
    float val = rsig[idx] * acc[r] * bet / sumk[idx];
    rwkv16[idx] = (_Float16)val;
  }
}

// ---------------------------------------------------------------------- launch

extern "C" void kernel_launch(void* const* d_in, const int* in_sizes, int n_in,
                              void* d_out, int out_size, void* d_ws, size_t ws_size,
                              hipStream_t stream) {
  const float* x     = (const float*)d_in[0];
  const float* tw    = (const float*)d_in[1];
  const float* alpha = (const float*)d_in[2];
  const float* beta  = (const float*)d_in[3];
  const float* gamma = (const float*)d_in[4];
  const float* Wk    = (const float*)d_in[5];
  const float* bk    = (const float*)d_in[6];
  const float* Wv    = (const float*)d_in[7];
  const float* bv    = (const float*)d_in[8];
  const float* Wr    = (const float*)d_in[9];
  const float* br    = (const float*)d_in[10];
  const float* Wo    = (const float*)d_in[11];
  const float* bo    = (const float*)d_in[12];
  float* out = (float*)d_out;

  const int BT = NB * NT;               // 8192 rows
  const int TOT = BT * NC;              // 8M elements

  char* ws = (char*)d_ws;
  size_t off = 0;
  auto take = [&](size_t bytes) -> void* {
    void* p = ws + off;
    off += (bytes + 255) & ~(size_t)255;
    return p;
  };
  _Float16* xs16   = (_Float16*)take((size_t)TOT * 2);
  _Float16* Wkt    = (_Float16*)take((size_t)NC * NC * 2);   // N-major
  _Float16* Wvt    = (_Float16*)take((size_t)NC * NC * 2);
  _Float16* Wrt    = (_Float16*)take((size_t)NC * NC * 2);
  _Float16* Wot    = (_Float16*)take((size_t)NC * NC * 2);
  float*    kbuf   = (float*)take((size_t)TOT * 4);          // -> sum_k after cumsum
  float*    vbuf   = (float*)take((size_t)TOT * 4);
  float*    rsig   = (float*)take((size_t)TOT * 4);
  __bf16*   kvT    = (__bf16*)take((size_t)TOT * 2);         // (B*NC, NT)
  _Float16* rwkv16 = (_Float16*)take((size_t)TOT * 2);
  __bf16*   Atiles = (__bf16*)take((size_t)NH * 64 * 32 * 16 * 2);  // 1 MB
  (void)ws_size; (void)in_sizes; (void)n_in; (void)out_size;

  // 1) time-shift + f16; weights -> N-major f16; decay fragment tiles
  shift_to_f16<<<TOT / 256, 256, 0, stream>>>(x, xs16, TOT);
  dim3 tgrid(NC / 32, NC / 32);
  transpose_w_f16<<<tgrid, 256, 0, stream>>>(Wk, Wkt, NC, NC);
  transpose_w_f16<<<tgrid, 256, 0, stream>>>(Wv, Wvt, NC, NC);
  transpose_w_f16<<<tgrid, 256, 0, stream>>>(Wr, Wrt, NC, NC);
  transpose_w_f16<<<tgrid, 256, 0, stream>>>(Wo, Wot, NC, NC);
  build_atiles<<<NH * 64, 32, 0, stream>>>(tw, Atiles);

  // 2) projections with fused activations
  dim3 ggrid(NC / 64, BT / 64);
  gemm_f16_wmma<<<ggrid, 256, 0, stream>>>(xs16, Wkt, bk, kbuf, nullptr, NC, 0);
  gemm_f16_wmma<<<ggrid, 256, 0, stream>>>(xs16, Wvt, bv, vbuf, nullptr, NC, 1);
  gemm_f16_wmma<<<ggrid, 256, 0, stream>>>(xs16, Wrt, br, rsig, nullptr, NC, 2);

  // 3) kv' transpose (alpha folded, bf16)    4) sum_k = cumsum(k) in place
  kvT_prep<<<dim3(NC / 32, NT / 32, NB), 256, 0, stream>>>(kbuf, vbuf, alpha, kvT);
  cumsum_t<<<(NB * NC) / 256, 256, 0, stream>>>(kbuf);

  // 5) causal decay matmul + fused rwkv epilogue
  wkv_wmma<<<NB * NH * (NT / 16), 128, 0, stream>>>(Atiles, beta, kvT, rsig, kbuf, rwkv16);

  // 6) output projection, scaled by gamma[t]
  gemm_f16_wmma<<<ggrid, 256, 0, stream>>>(rwkv16, Wot, bo, out, gamma, NC, 3);
}
